// NeuralSDE_6811818131923
// MI455X (gfx1250) — matrix operand order repaired and checked by reading, not compiled, not run on hardware
//
#include <hip/hip_runtime.h>
#include <hip/hip_bf16.h>

#define H 512
#define BATCH 262144
#define NSDE_EPS 1e-7f

typedef _Float16 half8  __attribute__((ext_vector_type(8)));
typedef _Float16 half16 __attribute__((ext_vector_type(16)));
typedef float    f32x8  __attribute__((ext_vector_type(8)));
typedef unsigned int u32x4 __attribute__((ext_vector_type(4)));
typedef int      i32x4 __attribute__((ext_vector_type(4)));
typedef int      i32x8 __attribute__((ext_vector_type(8)));

// LDS layout (bytes):
//   [0,       6144)   aLDS: [3][H] f32   (t*w1[0]+b1)
//   [6144,   12288)   cLDS: [3][H] f32   (w1[1])
//   [12288, 143360)   hLDS: [8 waves][16 rows][H] f16  (layer-1 A operand)
//   [143360,176128)   Bbuf: [2][16 cols][H] f16        (TDM-staged w2t tiles)
#define A_OFF   0
#define C_OFF   6144
#define HL_OFF  12288
#define B_OFF   143360
#define B_TILE_BYTES 16384
#define LDS_BYTES 176128

// ---------------------------------------------------------------------------
// One-time prep: w2 (f32, [K][N] row-major) -> w2t (f16, [N][K]) for 3 nets.
// ---------------------------------------------------------------------------
__global__ void nsde_transpose_w2(const float* __restrict__ pw2,
                                  const float* __restrict__ sw2,
                                  const float* __restrict__ rw2,
                                  _Float16* __restrict__ w2t) {
  __shared__ _Float16 tile[32][33];
  const float* src = (blockIdx.z == 0) ? pw2 : (blockIdx.z == 1 ? sw2 : rw2);
  const int k0 = blockIdx.x * 32, n0 = blockIdx.y * 32;
  for (int i = threadIdx.y; i < 32; i += 8)
    tile[i][threadIdx.x] = (_Float16)src[(size_t)(k0 + i) * H + (n0 + threadIdx.x)];
  __syncthreads();
  _Float16* dst = w2t + (size_t)blockIdx.z * H * H;
  for (int i = threadIdx.y; i < 32; i += 8)
    dst[(size_t)(n0 + i) * H + (k0 + threadIdx.x)] = tile[threadIdx.x][i];
}

// ---------------------------------------------------------------------------
// TDM: DMA one 16-row x 512-col f16 tile of w2t (row stride 512) into LDS.
// D# per CDNA5 ISA 08_async_tensor §8.3-8.5: count=1, type=2 ("image"),
// data_size=1 (2 bytes), tile_dim0=512 (contig K), tile_dim1=16 (rows),
// tensor_dim0=tensor_dim1=512, tensor_dim0_stride=512. Groups 2/3 zero (2D).
// ---------------------------------------------------------------------------
__device__ __forceinline__ void tdm_load_tile(const _Float16* gsrc,
                                              unsigned lds_byte_off) {
  unsigned long long ga = (unsigned long long)(const void*)gsrc;
  u32x4 g0;
  g0[0] = 1u;                                             // count=1, user mode
  g0[1] = lds_byte_off;                                   // lds_addr
  g0[2] = (unsigned)(ga & 0xffffffffu);                   // global_addr[31:0]
  g0[3] = (unsigned)((ga >> 32) & 0x01ffffffu) | 0x80000000u; // [56:32]|type=2
  i32x8 g1;
  g1[0] = 0x00010000;          // workgroup_mask=0, data_size=1 (2 bytes)
  g1[1] = (int)(512u << 16);   // tensor_dim0[15:0]=512 (atomic_barrier=0)
  g1[2] = (int)(512u << 16);   // tensor_dim0 hi=0, tensor_dim1[15:0]=512
  g1[3] = (int)(512u << 16);   // tensor_dim1 hi=0, tile_dim0=512
  g1[4] = 16;                  // tile_dim1=16, tile_dim2=0
  g1[5] = 512;                 // tensor_dim0_stride lo
  g1[6] = 0;                   // stride hi, tensor_dim1_stride lo (unused, 2D)
  g1[7] = 0;
  i32x4 gz = {0, 0, 0, 0};
#if defined(__clang_major__) && (__clang_major__ >= 23)
  i32x8 gz8 = {0, 0, 0, 0, 0, 0, 0, 0};
  __builtin_amdgcn_tensor_load_to_lds(g0, g1, gz, gz, gz8, 0);
#else
  __builtin_amdgcn_tensor_load_to_lds(g0, g1, gz, gz, 0);
#endif
}

// ---------------------------------------------------------------------------
// Fully fused NeuralSDE kernel. 256 threads = 8 wave32; each wave owns 16
// batch rows (one WMMA M-tile); workgroup covers 128 rows; grid = B/128.
// B tiles of w2t are TDM-staged into LDS once per workgroup (8x L2-traffic
// cut), double-buffered so the DMA overlaps WMMA compute.
// ---------------------------------------------------------------------------
__global__ __launch_bounds__(256)
void nsde_fused(const float* __restrict__ tptr, const float* __restrict__ x,
                const _Float16* __restrict__ w2t,
                const float* __restrict__ w1p, const float* __restrict__ b1p,
                const float* __restrict__ b2p, const float* __restrict__ w3p,
                const float* __restrict__ b3p,
                const float* __restrict__ w1s, const float* __restrict__ b1s,
                const float* __restrict__ b2s, const float* __restrict__ w3s,
                const float* __restrict__ b3s,
                const float* __restrict__ w1r, const float* __restrict__ b1r,
                const float* __restrict__ b2r, const float* __restrict__ w3r,
                const float* __restrict__ b3r,
                float* __restrict__ out) {
  extern __shared__ char smem[];
  float*    aLDS = (float*)(smem + A_OFF);
  float*    cLDS = (float*)(smem + C_OFF);
  _Float16* hLDS = (_Float16*)(smem + HL_OFF);
  _Float16* Bbuf = (_Float16*)(smem + B_OFF);   // [2][16][H] f16

  const int tid   = threadIdx.x;
  const int wave  = tid >> 5;
  const int lane  = tid & 31;
  const int nIdx  = lane & 15;   // WMMA B: lane -> N;  A: lane -> M
  const int kHalf = lane >> 4;   // which K half this lane holds

  const float* w1n[3] = {w1p, w1s, w1r};
  const float* b1n[3] = {b1p, b1s, b1r};
  const float* b2n[3] = {b2p, b2s, b2r};
  const float* w3n[3] = {w3p, w3s, w3r};
  const float* b3n[3] = {b3p, b3s, b3r};

  const float t0 = tptr[0];

  // Per-network layer-1 constant vectors a, c (shared by all waves).
  for (int idx = tid; idx < 3 * H; idx += 256) {
    const int net = idx >> 9;
    const int j   = idx & (H - 1);
    aLDS[idx] = fmaf(t0, w1n[net][j], b1n[net][j]);
    cLDS[idx] = w1n[net][H + j];
  }

  const int row0 = blockIdx.x * 128 + wave * 16;
  const float xr = x[(size_t)(row0 + nIdx) * 2];   // x[:,0] for this wave's rows
  _Float16* hW = hLDS + (size_t)wave * 16 * H;

  __syncthreads();

  float fres[8], gres[8], hres[8];

  #pragma unroll
  for (int net = 0; net < 3; ++net) {
    const bool isSoftmax = (net != 1);
    const _Float16* Bnet = w2t + (size_t)net * H * H;

    // Kick off the TDM for this net's first B tile; it streams while we
    // compute layer 1 below. (Buffer 0 is guaranteed idle here.)
    if (wave == 0) tdm_load_tile(Bnet, B_OFF);

    // ---- layer 1: h1[r,:] = softmax(a + x_r * c) over H, stored f16 in LDS --
    float av[16], cv[16];
    #pragma unroll
    for (int c = 0; c < 16; ++c) {
      av[c] = aLDS[net * H + c * 32 + lane];
      cv[c] = cLDS[net * H + c * 32 + lane];
    }
    for (int r = 0; r < 16; ++r) {
      const float xv = __shfl(xr, r, 32);
      float z[16];
      float m = -3.0e38f;
      #pragma unroll
      for (int c = 0; c < 16; ++c) { z[c] = fmaf(xv, cv[c], av[c]); m = fmaxf(m, z[c]); }
      #pragma unroll
      for (int off = 16; off >= 1; off >>= 1) m = fmaxf(m, __shfl_xor(m, off, 32));
      float d = 0.f;
      #pragma unroll
      for (int c = 0; c < 16; ++c) { z[c] = __expf(z[c] - m); d += z[c]; }
      #pragma unroll
      for (int off = 16; off >= 1; off >>= 1) d += __shfl_xor(d, off, 32);
      const float inv = 1.0f / d;
      _Float16* hRow = hW + r * H;
      #pragma unroll
      for (int c = 0; c < 16; ++c) hRow[c * 32 + lane] = (_Float16)(z[c] * inv);
    }
    // Same-wave LDS ops are in-order: h1 stores are visible to ds_loads below.

    // ---- layer 2+3 fused: WMMA over K per N-tile, online activation+w3 ------
    float mAcc[8], dAcc[8], sAcc[8];
    #pragma unroll
    for (int j = 0; j < 8; ++j) { mAcc[j] = -3.0e38f; dAcc[j] = 0.f; sAcc[j] = 0.f; }

    // A frag (16x32 f16): lanes -> M, two contiguous 8-half LDS loads per tile.
    const _Float16* Arow = hW + nIdx * H + kHalf * 16;

    for (int nt = 0; nt < 32; ++nt) {
      // Double-buffered TDM pipeline: issue tile nt+1, wait for tile nt.
      if (wave == 0) {
        if (nt + 1 < 32) {
          tdm_load_tile(Bnet + (size_t)(nt + 1) * 16 * H,
                        B_OFF + (unsigned)((nt + 1) & 1) * B_TILE_BYTES);
          __builtin_amdgcn_s_wait_tensorcnt(1);
        } else {
          __builtin_amdgcn_s_wait_tensorcnt(0);
        }
      }
      __syncthreads();   // tile nt resident in Bbuf[nt&1] for all waves

      const int n = nt * 16 + nIdx;
      const float w3v = w3n[net][n];
      const float b2v = b2n[net][n];
      // B frag (32x16 f16): lane -> N col, 16 contiguous K at kHalf*16.
      const _Float16* Bcol = Bbuf + (nt & 1) * (16 * H) + nIdx * H + kHalf * 16;

      // Two independent accumulator chains (k-tiles 0-7 and 8-15) keep the
      // XDL pipe fed despite the WMMA C->C RAW dependency.
      f32x8 acc0 = {}, acc1 = {};
      #pragma unroll 4
      for (int kt = 0; kt < 8; ++kt) {
        half8 a0l = *(const half8*)(Arow + kt * 32);
        half8 a0h = *(const half8*)(Arow + kt * 32 + 8);
        half8 a1l = *(const half8*)(Arow + (kt + 8) * 32);
        half8 a1h = *(const half8*)(Arow + (kt + 8) * 32 + 8);
        half8 b0l = *(const half8*)(Bcol + kt * 32);
        half8 b0h = *(const half8*)(Bcol + kt * 32 + 8);
        half8 b1l = *(const half8*)(Bcol + (kt + 8) * 32);
        half8 b1h = *(const half8*)(Bcol + (kt + 8) * 32 + 8);
        half16 a0 = __builtin_shufflevector(a0l, a0h, 0,1,2,3,4,5,6,7,8,9,10,11,12,13,14,15);
        half16 a1 = __builtin_shufflevector(a1l, a1h, 0,1,2,3,4,5,6,7,8,9,10,11,12,13,14,15);
        half16 b0 = __builtin_shufflevector(b0l, b0h, 0,1,2,3,4,5,6,7,8,9,10,11,12,13,14,15);
        half16 b1 = __builtin_shufflevector(b1l, b1h, 0,1,2,3,4,5,6,7,8,9,10,11,12,13,14,15);
        acc0 = __builtin_amdgcn_wmma_f32_16x16x32_f16(
            false, a0, false, b0, (short)0, acc0, false, false);
        acc1 = __builtin_amdgcn_wmma_f32_16x16x32_f16(
            false, a1, false, b1, (short)0, acc1, false, false);
      }

      // C frag: VGPR j, lanes 0-15 -> row j, lanes 16-31 -> row j+8; col=nIdx.
      if (isSoftmax) {
        #pragma unroll
        for (int j = 0; j < 8; ++j) {
          const float zv = acc0[j] + acc1[j] + b2v;
          const float nm = fmaxf(mAcc[j], zv);
          const float sc = __expf(mAcc[j] - nm);
          const float e  = __expf(zv - nm);
          dAcc[j] = dAcc[j] * sc + e;
          sAcc[j] = sAcc[j] * sc + e * w3v;
          mAcc[j] = nm;
        }
      } else {
        #pragma unroll
        for (int j = 0; j < 8; ++j) {
          const float zv = acc0[j] + acc1[j] + b2v;
          const float sp = (zv > 20.f) ? zv : __logf(1.f + __expf(zv));
          sAcc[j] += sp * w3v;
        }
      }
      __syncthreads();   // all waves done reading Bbuf[nt&1] before reuse
    }

    // ---- combine across the 16 column-lanes (stays inside each 16-lane half)
    const float b3v = b3n[net][0];
    #pragma unroll
    for (int j = 0; j < 8; ++j) {
      float m = mAcc[j], d = dAcc[j], s = sAcc[j];
      if (isSoftmax) {
        #pragma unroll
        for (int mask = 1; mask <= 8; mask <<= 1) {
          const float om = __shfl_xor(m, mask, 32);
          const float od = __shfl_xor(d, mask, 32);
          const float os = __shfl_xor(s, mask, 32);
          const float nm = fmaxf(m, om);
          const float ea = __expf(m - nm), eb = __expf(om - nm);
          d = d * ea + od * eb;
          s = s * ea + os * eb;
          m = nm;
        }
        s = s / d;
      } else {
        #pragma unroll
        for (int mask = 1; mask <= 8; mask <<= 1) s += __shfl_xor(s, mask, 32);
      }
      const float res = s + b3v;
      if (net == 0)      fres[j] = res;
      else if (net == 1) gres[j] = res;
      else               hres[j] = res;
    }
  }

  // ---- u = (f-h)/g_safe ; kl = 0.5 u^2 ; out = [f, kl] ----------------------
  if (nIdx == 0) {  // lanes 0 (rows j) and 16 (rows j+8) hold the results
    #pragma unroll
    for (int j = 0; j < 8; ++j) {
      const float g   = gres[j];
      const float sgn = (g > 0.f) ? 1.f : ((g < 0.f) ? -1.f : 0.f);
      const float gs  = (fabsf(g) > NSDE_EPS) ? g : sgn * NSDE_EPS;
      const float u   = (fres[j] - hres[j]) / gs;
      const float kl  = 0.5f * u * u;
      const int row   = row0 + j + kHalf * 8;
      *(float2*)(out + (size_t)row * 2) = make_float2(fres[j], kl);
    }
  }
}

extern "C" void kernel_launch(void* const* d_in, const int* in_sizes, int n_in,
                              void* d_out, int out_size, void* d_ws, size_t ws_size,
                              hipStream_t stream) {
  (void)in_sizes; (void)n_in; (void)out_size; (void)ws_size;
  const float* t   = (const float*)d_in[0];
  const float* x   = (const float*)d_in[1];
  const float* pw1 = (const float*)d_in[2];
  const float* pb1 = (const float*)d_in[3];
  const float* pw2 = (const float*)d_in[4];
  const float* pb2 = (const float*)d_in[5];
  const float* pw3 = (const float*)d_in[6];
  const float* pb3 = (const float*)d_in[7];
  const float* sw1 = (const float*)d_in[8];
  const float* sb1 = (const float*)d_in[9];
  const float* sw2 = (const float*)d_in[10];
  const float* sb2 = (const float*)d_in[11];
  const float* sw3 = (const float*)d_in[12];
  const float* sb3 = (const float*)d_in[13];
  const float* rw1 = (const float*)d_in[14];
  const float* rb1 = (const float*)d_in[15];
  const float* rw2 = (const float*)d_in[16];
  const float* rb2 = (const float*)d_in[17];
  const float* rw3 = (const float*)d_in[18];
  const float* rb3 = (const float*)d_in[19];
  float* out = (float*)d_out;

  _Float16* w2t = (_Float16*)d_ws;  // 3 * 512 * 512 f16 = 1.5 MB

  nsde_transpose_w2<<<dim3(H / 32, H / 32, 3), dim3(32, 8), 0, stream>>>(
      pw2, sw2, rw2, w2t);

  nsde_fused<<<BATCH / 128, 256, LDS_BYTES, stream>>>(
      t, x, w2t,
      pw1, pb1, pb2, pw3, pb3,
      sw1, sb1, sb2, sw3, sb3,
      rw1, rb1, rb2, rw3, rb3,
      out);
}
